// EGCL_58660663329297
// MI455X (gfx1250) — matrix operand order repaired and checked by reading, not compiled, not run on hardware
//
#include <hip/hip_runtime.h>

typedef unsigned int u32;
typedef unsigned short u16;
typedef __attribute__((ext_vector_type(16))) __bf16 v16bf;
typedef __attribute__((ext_vector_type(8)))  float  v8f;

#define NNODES 768
#define NEDGE  767
#define NTILES 48

// ---- LDS layout (dword offsets) ----
#define OFF_W0    0        // we0 packed bf16 tiles: 8kc x 8nt x 256dw = 16384
#define OFF_W1    16384    // 4kc x 8nt x 256 = 8192
#define OFF_W2    24576
#define OFF_WX0   32768
#define OFF_WX1   40960
#define OFF_A256  49152    // [16][256] bf16 = 2048 dw
#define OFF_B1    51200    // [16][128] bf16 = 1024 dw
#define OFF_B2    52224    // [16][128] bf16 = 1024 dw
#define OFF_MF    53248    // [16][128] f32  = 2048 dw (also reused as phi_h scratch)
#define OFF_HR    55296    // 128 f32
#define OFF_BE0   55424
#define OFF_BE1   55552
#define OFF_BE2   55680
#define OFF_BX0   55808
#define OFF_BX1   55936
#define OFF_W0R0  56064
#define OFF_W0R1  56192
#define OFF_WXL   56320    // 256
#define OFF_WINF  56576    // 128
#define OFF_LEN2  56704    // 32
#define OFF_LENS  56736    // 32
#define OFF_VEC   56768    // 96
#define OFF_GATE  56864    // 16
#define OFF_MI    56880    // 128
#define OFF_SSCR  57008    // 96
#define OFF_MISC  57104    // 4 : bxl0, bxl1, binf
#define TOTAL_DW  57108
#define SMEM_BYTES (TOTAL_DW * 4)

static __device__ __forceinline__ u16 f2bf(float f) {
    u32 u = __builtin_bit_cast(u32, f);
    u32 r = (u + 0x7FFFu + ((u >> 16) & 1u)) >> 16;
    return (u16)r;
}
static __device__ __forceinline__ float bf2f(u16 h) {
    return __builtin_bit_cast(float, ((u32)h) << 16);
}
static __device__ __forceinline__ u32 packbf(float a, float b) {
    return (u32)f2bf(a) | ((u32)f2bf(b) << 16);
}
static __device__ __forceinline__ float silu_f(float x) {
    return x / (1.0f + __expf(-x));
}

union FragU { u32 u[8]; v16bf v; };

// GEMM over one 16-wide N slice: A from LDS (row-major bf16 pairs, rowDW dwords/row),
// B from LDS packed fragment tiles (kc*8+nt)*256 + lane*8. K = nkc*32.
static __device__ __forceinline__ v8f gemm_bf16(const u32* act, int rowDW,
                                                const u32* wp, int nkc,
                                                int nt, int lane, v8f c) {
    const int m = lane & 15, g = lane >> 4;
#pragma unroll
    for (int kc = 0; kc < nkc; ++kc) {
        FragU A, B;
        const u32* ar = act + m * rowDW + kc * 16 + 4 * g;   // A layout: lanes0-15 K0-7, lanes16-31 K8-15, vgpr4-7 +16
        *(uint4*)&A.u[0] = *(const uint4*)(ar);
        *(uint4*)&A.u[4] = *(const uint4*)(ar + 8);
        const u32* br = wp + (kc * 8 + nt) * 256 + lane * 8;
        *(uint4*)&B.u[0] = *(const uint4*)(br);
        *(uint4*)&B.u[4] = *(const uint4*)(br + 4);
        c = __builtin_amdgcn_wmma_f32_16x16x32_bf16(false, A.v, false, B.v,
                                                    (short)0, c, false, false);
    }
    return c;
}

// Pack f32 [K x 128] weight (starting at row `rowoff`) into bf16 B-fragment tiles in LDS.
static __device__ void stage_pack(u32* dst, const float* src, int kcn, int rowoff, int tid) {
    const int ndw = kcn * 8 * 256;
    for (int d = tid; d < ndw; d += 256) {
        int tile = d >> 8, w = d & 255;
        int l = w >> 3, v = w & 7;
        int kc = tile >> 3, nt = tile & 7;
        int k = kc * 32 + ((l >> 4) << 3) + ((v >> 2) << 4) + 2 * (v & 3) + rowoff;  // lanes16-31 carry K+8, v4-7 +16
        int n = nt * 16 + (l & 15);
        dst[d] = packbf(src[k * 128 + n], src[(k + 1) * 128 + n]);
    }
}

// store C tile (16x16 f32, ISA C layout) as bf16 row-major [16][128] (+optional f32 copy)
static __device__ __forceinline__ void store_act(u16* dstu, float* mf, v8f c,
                                                 int nt, int lane, bool dosilu) {
    const int nl = lane & 15, hi = lane >> 4;
#pragma unroll
    for (int g = 0; g < 8; ++g) {
        float x = c[g];
        if (dosilu) x = silu_f(x);
        int M = g + 8 * hi, Nc = nt * 16 + nl;
        dstu[M * 128 + Nc] = f2bf(x);
        if (mf) mf[M * 128 + Nc] = x;
    }
}

__global__ __launch_bounds__(256, 1)
void egcl_kernel(const float* __restrict__ nvec, const float* __restrict__ nfeat,
                 const float* __restrict__ we0, const float* __restrict__ be0,
                 const float* __restrict__ we1, const float* __restrict__ be1,
                 const float* __restrict__ we2, const float* __restrict__ be2,
                 const float* __restrict__ wx0, const float* __restrict__ bx0,
                 const float* __restrict__ wx1, const float* __restrict__ bx1,
                 const float* __restrict__ wxl, const float* __restrict__ bxl,
                 const float* __restrict__ winf, const float* __restrict__ binf,
                 const float* __restrict__ wh0, const float* __restrict__ bh0,
                 const float* __restrict__ wh1, const float* __restrict__ bh1,
                 const float* __restrict__ whl, const float* __restrict__ bhl,
                 float* __restrict__ out) {
    extern __shared__ __align__(16) char smem_raw[];
    u32* S = (u32*)smem_raw;

    u32* W0  = S + OFF_W0;  u32* W1  = S + OFF_W1;  u32* W2 = S + OFF_W2;
    u32* WX0 = S + OFF_WX0; u32* WX1 = S + OFF_WX1;
    u32* A256 = S + OFF_A256;
    u32* B1 = S + OFF_B1;   u32* B2 = S + OFF_B2;
    u16* B1u = (u16*)B1;    u16* B2u = (u16*)B2;
    float* MF   = (float*)(S + OFF_MF);
    float* HR   = (float*)(S + OFF_HR);
    float* BE0s = (float*)(S + OFF_BE0); float* BE1s = (float*)(S + OFF_BE1);
    float* BE2s = (float*)(S + OFF_BE2); float* BX0s = (float*)(S + OFF_BX0);
    float* BX1s = (float*)(S + OFF_BX1);
    float* W0R0 = (float*)(S + OFF_W0R0); float* W0R1 = (float*)(S + OFF_W0R1);
    float* WXLs = (float*)(S + OFF_WXL);  float* WINFs = (float*)(S + OFF_WINF);
    float* LEN2 = (float*)(S + OFF_LEN2); float* LENS = (float*)(S + OFF_LENS);
    float* VEC  = (float*)(S + OFF_VEC);  float* GATE = (float*)(S + OFF_GATE);
    float* MI   = (float*)(S + OFF_MI);   float* SSCR = (float*)(S + OFF_SSCR);
    float* MISC = (float*)(S + OFF_MISC);

    const int r    = blockIdx.x;
    const int tid  = threadIdx.x;
    const int lane = tid & 31;
    const int wave = tid >> 5;
    const int nt   = wave;                 // each wave owns one 16-wide N slice
    const int nl   = lane & 15, hi = lane >> 4;
    const int Nc   = nt * 16 + nl;

    // ---- stage weights (bf16 fragment layout) + misc into LDS ----
    stage_pack(W0,  we0, 8, 2, tid);       // we0 rows 2..257 (h_s ++ h_r part)
    stage_pack(W1,  we1, 4, 0, tid);
    stage_pack(W2,  we2, 4, 0, tid);
    stage_pack(WX0, wx0, 4, 0, tid);
    stage_pack(WX1, wx1, 4, 0, tid);
    for (int i = tid; i < 128; i += 256) {
        HR[i]   = nfeat[r * 128 + i];
        BE0s[i] = be0[i]; BE1s[i] = be1[i]; BE2s[i] = be2[i];
        BX0s[i] = bx0[i]; BX1s[i] = bx1[i];
        W0R0[i] = we0[i]; W0R1[i] = we0[128 + i];
        WXLs[2 * i] = wxl[2 * i]; WXLs[2 * i + 1] = wxl[2 * i + 1];
        WINFs[i] = winf[i];
        MI[i] = 0.0f;
    }
    if (tid == 0) { MISC[0] = bxl[0]; MISC[1] = bxl[1]; MISC[2] = binf[0]; }
    // A-matrix columns 128..255 (= h_r) are constant for the whole block
    {
        const float* hrp = nfeat + r * 128;
        for (int d = tid; d < 16 * 64; d += 256) {
            int m = d >> 6, i = d & 63;
            A256[m * 128 + 64 + i] = packbf(hrp[2 * i], hrp[2 * i + 1]);
        }
    }
    __syncthreads();

    float sa0 = 0.0f, sa1 = 0.0f, sa2 = 0.0f;  // shift partials (threads 0..31)

    for (int t = 0; t < NTILES; ++t) {
        // ---- per-edge geometry (16 edges) ----
        if (tid < 16) {
            int j = t * 16 + tid;
            int s = (j < NEDGE) ? (j + ((j >= r) ? 1 : 0)) : r;  // pad edge -> self (vec=0)
            float d[6];
#pragma unroll
            for (int q = 0; q < 6; ++q) d[q] = nvec[r * 6 + q] - nvec[s * 6 + q];
#pragma unroll
            for (int v = 0; v < 2; ++v) {
                float l2 = d[v*3]*d[v*3] + d[v*3+1]*d[v*3+1] + d[v*3+2]*d[v*3+2];
                l2 = fmaxf(l2, 1e-20f);
                LEN2[tid * 2 + v] = l2;
                LENS[tid * 2 + v] = sqrtf(l2);
                VEC[tid*6 + v*3 + 0] = d[v*3 + 0];
                VEC[tid*6 + v*3 + 1] = d[v*3 + 1];
                VEC[tid*6 + v*3 + 2] = d[v*3 + 2];
            }
        }
        // ---- fill A columns 0..127 with h[sender] ----
        {
            int m = tid >> 4;
            int j = t * 16 + m;
            int s = (j < NEDGE) ? (j + ((j >= r) ? 1 : 0)) : r;
            const float* hs = nfeat + s * 128;
            int db = (tid & 15) * 4;
#pragma unroll
            for (int i = 0; i < 4; ++i) {
                int k = (db + i) * 2;
                A256[m * 128 + db + i] = packbf(hs[k], hs[k + 1]);
            }
        }
        __syncthreads();

        // ---- phi_e layer0: C init = be0 + lengths^2 @ we0[0:2]  (rank-2 VALU) ----
        v8f c;
#pragma unroll
        for (int g = 0; g < 8; ++g) {
            int M = g + 8 * hi;
            c[g] = BE0s[Nc] + LEN2[M * 2 + 0] * W0R0[Nc] + LEN2[M * 2 + 1] * W0R1[Nc];
        }
        c = gemm_bf16(A256, 128, W0, 8, nt, lane, c);
        store_act(B1u, nullptr, c, nt, lane, true);          // silu
        __syncthreads();

        // ---- phi_e layer1 ----
#pragma unroll
        for (int g = 0; g < 8; ++g) c[g] = BE1s[Nc];
        c = gemm_bf16(B1, 64, W1, 4, nt, lane, c);
        store_act(B2u, nullptr, c, nt, lane, true);          // silu
        __syncthreads();

        // ---- phi_e layer2 -> m_ij (linear); keep f32 + bf16 copies ----
#pragma unroll
        for (int g = 0; g < 8; ++g) c[g] = BE2s[Nc];
        c = gemm_bf16(B2, 64, W2, 4, nt, lane, c);
        store_act(B1u, MF, c, nt, lane, false);
        __syncthreads();

        // ---- phi_x layer0 ----
#pragma unroll
        for (int g = 0; g < 8; ++g) c[g] = BX0s[Nc];
        c = gemm_bf16(B1, 64, WX0, 4, nt, lane, c);
        store_act(B2u, nullptr, c, nt, lane, true);          // silu
        __syncthreads();

        // ---- phi_x layer1 (activate_final=True) ----
#pragma unroll
        for (int g = 0; g < 8; ++g) c[g] = BX1s[Nc];
        c = gemm_bf16(B2, 64, WX1, 4, nt, lane, c);
        store_act(B1u, nullptr, c, nt, lane, true);          // silu
        __syncthreads();

        // ---- px -> shift partials (wave0) ; gate (wave1 lanes<16) ----
        if (tid < 32) {
            int me = tid >> 1, vv = tid & 1;
            float acc = MISC[vv];                            // bxl
            for (int k = 0; k < 128; ++k)
                acc += bf2f(B1u[me * 128 + k]) * WXLs[k * 2 + vv];
            float sc = acc / (1.0f + LENS[me * 2 + vv]);     // C_NORM = 1
            sa0 += sc * VEC[me * 6 + vv * 3 + 0];            // padded edge has VEC=0
            sa1 += sc * VEC[me * 6 + vv * 3 + 1];
            sa2 += sc * VEC[me * 6 + vv * 3 + 2];
        } else if (tid < 48) {
            int m = tid - 32;
            int j = t * 16 + m;
            float acc = MISC[2];                             // binf
            for (int k = 0; k < 128; ++k) acc += MF[m * 128 + k] * WINFs[k];
            float g = 1.0f / (1.0f + __expf(-acc));
            GATE[m] = (j < NEDGE) ? g : 0.0f;                // mask padded edge
        }
        __syncthreads();

        // ---- m_i accumulation ----
        if (tid < 128) {
            float a = MI[tid];
            for (int m = 0; m < 16; ++m) a += MF[m * 128 + tid] * GATE[m];
            MI[tid] = a;
        }
        __syncthreads();
    }

    // ---- finalize ----
    if (tid < 32) { SSCR[tid*3] = sa0; SSCR[tid*3+1] = sa1; SSCR[tid*3+2] = sa2; }
    if (tid < 128) MI[tid] = MI[tid] * (1.0f / sqrtf(767.0f));
    __syncthreads();

    if (tid < 6) {
        int vv = tid / 3, k = tid % 3;
        float s = 0.0f;
        for (int m = 0; m < 16; ++m) s += SSCR[(m * 2 + vv) * 3 + k];
        out[r * 6 + vv * 3 + k] = nvec[r * 6 + vv * 3 + k] + s * (1.0f / 767.0f);
    }

    // phi_h (per-node matvec chain) ; scratch reuses MF
    float* PH1 = MF;
    float* PH2 = MF + 128;
    if (tid < 128) {
        float a = bh0[tid];
        for (int k = 0; k < 128; ++k) a += MI[k] * wh0[k * 128 + tid];
        for (int k = 0; k < 128; ++k) a += HR[k] * wh0[(128 + k) * 128 + tid];
        PH1[tid] = silu_f(a);
    }
    __syncthreads();
    if (tid < 128) {
        float a = bh1[tid];
        for (int k = 0; k < 128; ++k) a += PH1[k] * wh1[k * 128 + tid];
        PH2[tid] = silu_f(a);
    }
    __syncthreads();
    if (tid < 128) {
        float a = bhl[tid];
        for (int k = 0; k < 128; ++k) a += PH2[k] * whl[k * 128 + tid];
        out[NNODES * 6 + r * 128 + tid] = a + HR[tid];       // residual_h
    }
}

extern "C" void kernel_launch(void* const* d_in, const int* in_sizes, int n_in,
                              void* d_out, int out_size, void* d_ws, size_t ws_size,
                              hipStream_t stream) {
    (void)in_sizes; (void)n_in; (void)d_ws; (void)ws_size; (void)out_size;
    const float* nvec  = (const float*)d_in[0];
    const float* nfeat = (const float*)d_in[1];
    const float* we0 = (const float*)d_in[2];  const float* be0 = (const float*)d_in[3];
    const float* we1 = (const float*)d_in[4];  const float* be1 = (const float*)d_in[5];
    const float* we2 = (const float*)d_in[6];  const float* be2 = (const float*)d_in[7];
    const float* wx0 = (const float*)d_in[8];  const float* bx0 = (const float*)d_in[9];
    const float* wx1 = (const float*)d_in[10]; const float* bx1 = (const float*)d_in[11];
    const float* wxl = (const float*)d_in[12]; const float* bxl = (const float*)d_in[13];
    const float* winf = (const float*)d_in[14]; const float* binf = (const float*)d_in[15];
    const float* wh0 = (const float*)d_in[16]; const float* bh0 = (const float*)d_in[17];
    const float* wh1 = (const float*)d_in[18]; const float* bh1 = (const float*)d_in[19];
    const float* whl = (const float*)d_in[20]; const float* bhl = (const float*)d_in[21];
    float* out = (float*)d_out;

    hipFuncSetAttribute((const void*)egcl_kernel,
                        hipFuncAttributeMaxDynamicSharedMemorySize, SMEM_BYTES);
    egcl_kernel<<<NNODES, 256, SMEM_BYTES, stream>>>(
        nvec, nfeat, we0, be0, we1, be1, we2, be2,
        wx0, bx0, wx1, bx1, wxl, bxl, winf, binf,
        wh0, bh0, wh1, bh1, whl, bhl, out);
}
